// TripletLoss_52407190946022
// MI455X (gfx1250) — compile-verified, matrix operand order checked by reading, not run on hardware
//
#include <hip/hip_runtime.h>
#include <hip/hip_bf16.h>

#define ALPHA 0.2f
#define B 512
#define D 128

typedef __attribute__((ext_vector_type(2)))  float        v2f;
typedef __attribute__((ext_vector_type(8)))  float        v8f;
typedef __attribute__((ext_vector_type(16))) __bf16       v16bf;

// ---------------------------------------------------------------------------
// Kernel 1: Gram matrix via WMMA.  d[i][j] = 2 - 2 * dot(x_i, x_j)
// One wave32 per 16x16 tile of the 512x512 output.  K=128 reduced in the
// matrix pipe.
// ---------------------------------------------------------------------------
__global__ __launch_bounds__(32)
void gram_wmma_kernel(const float* __restrict__ x, float* __restrict__ d) {
  const int ti   = blockIdx.x;       // row tile (0..31)
  const int tj   = blockIdx.y;       // col tile (0..31)
  const int lane = threadIdx.x;      // 0..31 (wave32)
  const int half = lane >> 4;        // 0 or 1
  const int l16  = lane & 15;

  v8f c = {};

#if __has_builtin(__builtin_amdgcn_wmma_f32_16x16x4_f32)
  // f32 WMMA 16x16x4.  A-matrix (16x4) layout: lanes 0-15 hold M=0..15 with
  // VGPR0=K0, VGPR1=K1; lanes 16-31 hold VGPR0=K2, VGPR1=K3.  So each lane's
  // fragment is a contiguous float2 at column k0 + 2*half of its row.  The
  // B-matrix (4x16) mirrors this (K split across lane halves), and since
  // B[k][n] = X[tj*16+n][k], the load pattern is identical with tj.
  const float* arow = x + (size_t)(ti * 16 + l16) * D + 2 * half;
  const float* brow = x + (size_t)(tj * 16 + l16) * D + 2 * half;
  #pragma unroll
  for (int k0 = 0; k0 < D; k0 += 4) {
    v2f a = *(const v2f*)(arow + k0);
    v2f b = *(const v2f*)(brow + k0);
    c = __builtin_amdgcn_wmma_f32_16x16x4_f32(
        /*neg_a=*/false, a, /*neg_b=*/false, b,
        /*c_mod=*/(short)0, c, /*reuse_a=*/false, /*reuse_b=*/false);
  }
#else
  // Fallback: bf16 WMMA 16x16x32 (codegen-confirmed builtin).
  // 16-bit A layout: lane l (half=l/16) holds K in [8*half, 8*half+8) for
  // elements 0..7 and [16+8*half, 16+8*half+8) for elements 8..15.
  const float* arow = x + (size_t)(ti * 16 + l16) * D;
  const float* brow = x + (size_t)(tj * 16 + l16) * D;
  #pragma unroll
  for (int k0 = 0; k0 < D; k0 += 32) {
    v16bf a, b;
    #pragma unroll
    for (int e = 0; e < 8; ++e) {
      a[e]     = (__bf16)arow[k0 + 8 * half + e];
      a[e + 8] = (__bf16)arow[k0 + 16 + 8 * half + e];
      b[e]     = (__bf16)brow[k0 + 8 * half + e];
      b[e + 8] = (__bf16)brow[k0 + 16 + 8 * half + e];
    }
    c = __builtin_amdgcn_wmma_f32_16x16x32_bf16(
        false, a, false, b, (short)0, c, false, false);
  }
#endif

  // C/D layout: VGPR i -> (M = i + 8*half, N = l16).  Store d = 2 - 2*G.
  const int col = tj * 16 + l16;
  #pragma unroll
  for (int i = 0; i < 8; ++i) {
    const int row = ti * 16 + i + 8 * half;
    d[(size_t)row * B + col] = 2.0f - 2.0f * c[i];
  }
}

// ---------------------------------------------------------------------------
// Kernel 2: semi-hard mining.  One block (8 waves) per anchor a.
// For each valid (a,p) pair, lanes scan negatives n; candidate test is
// 0 < d_ap - d_an + alpha < alpha and label(n) != label(a).  Among candidates
// pick argmax of noise[a,p,:] (first occurrence on ties, matching jnp.argmax),
// accumulate L[a,p,n*].  Deterministic per-anchor partial sum (no f32 atomics).
// ---------------------------------------------------------------------------
__global__ __launch_bounds__(256)
void mine_kernel(const float* __restrict__ d, const int* __restrict__ target,
                 const float* __restrict__ noise, float* __restrict__ partial) {
  __shared__ float dn[B];
  __shared__ int   tg[B];
  __shared__ float wsum[8];

  const int a = blockIdx.x;
  const int t = threadIdx.x;

  for (int i = t; i < B; i += 256) {
    dn[i] = d[(size_t)a * B + i];
    tg[i] = target[i];
  }
  __syncthreads();

  const int ta   = tg[a];
  const int lane = t & 31;
  const int wave = t >> 5;

  const float* nrow_base = noise + (size_t)a * B * B;
  float acc = 0.0f;

  for (int p = wave; p < B; p += 8) {         // wave-uniform positive index
    if (p == a || tg[p] != ta) continue;      // pair_mask (uniform skip)

    const float  dap = dn[p];
    const float* nr  = nrow_base + (size_t)p * B;

    float best_s = -1.0f;
    int   best_n = 0;
    for (int n = lane; n < B; n += 32) {
      const float L = dap - dn[n] + ALPHA;
      if (L > 0.0f && L < ALPHA && tg[n] != ta) {
        const float s = nr[n];                // touch noise only at candidates
        if (s > best_s) { best_s = s; best_n = n; }  // first-occurrence ties
      }
    }
    // wave32 argmax reduction, lower index wins ties
    #pragma unroll
    for (int off = 16; off >= 1; off >>= 1) {
      const float os = __shfl_xor(best_s, off, 32);
      const int   on = __shfl_xor(best_n, off, 32);
      if (os > best_s || (os == best_s && on < best_n)) {
        best_s = os; best_n = on;
      }
    }
    if (lane == 0 && best_s >= 0.0f) {        // has_cand (noise in [0,1))
      const float tl = dap - dn[best_n] + ALPHA;
      acc += fmaxf(tl, 0.0f);
    }
  }

  if (lane == 0) wsum[wave] = acc;
  __syncthreads();
  if (t == 0) {
    float s = 0.0f;
    #pragma unroll
    for (int i = 0; i < 8; ++i) s += wsum[i];
    partial[a] = s;
  }
}

// ---------------------------------------------------------------------------
// Kernel 3: deterministic tree reduction of 512 partials -> scalar.
// ---------------------------------------------------------------------------
__global__ __launch_bounds__(512)
void reduce_kernel(const float* __restrict__ partial, float* __restrict__ out) {
  __shared__ float sh[B];
  const int t = threadIdx.x;
  sh[t] = partial[t];
  __syncthreads();
  for (int s = 256; s >= 1; s >>= 1) {
    if (t < s) sh[t] += sh[t + s];
    __syncthreads();
  }
  if (t == 0) out[0] = sh[0];
}

// ---------------------------------------------------------------------------
extern "C" void kernel_launch(void* const* d_in, const int* in_sizes, int n_in,
                              void* d_out, int out_size, void* d_ws, size_t ws_size,
                              hipStream_t stream) {
  const float* x      = (const float*)d_in[0];   // [512,128] f32 unit embeddings
  const int*   target = (const int*)d_in[1];     // [512] i32 labels
  const float* noise  = (const float*)d_in[2];   // [512,512,512] f32
  float*       out    = (float*)d_out;           // scalar

  float* dmat    = (float*)d_ws;                 // 512*512 f32 = 1 MB
  float* partial = dmat + (size_t)B * B;         // 512 f32

  gram_wmma_kernel<<<dim3(B / 16, B / 16), 32, 0, stream>>>(x, dmat);
  mine_kernel<<<B, 256, 0, stream>>>(dmat, target, noise, partial);
  reduce_kernel<<<1, B, 0, stream>>>(partial, out);
}
